// HybridMixerLayer_13443247637022
// MI455X (gfx1250) — compile-verified, matrix operand order checked by reading, not run on hardware
//
#include <hip/hip_runtime.h>
#include <hip/hip_bf16.h>
#include <math.h>

// ---------------------------------------------------------------------------
// Problem constants (from reference)
// ---------------------------------------------------------------------------
#define B_    64
#define L_    32
#define D_    1024
#define DFF_  2048
#define BLK_  128
#define N_    256
#define BASIS_ 8
#define BLD_  ((size_t)B_ * L_ * D_)   // 2,097,152

typedef __attribute__((ext_vector_type(16))) __bf16 v16bf;
typedef __attribute__((ext_vector_type(8)))  float  v8f;

// ---------------------------------------------------------------------------
// WMMA 16-bit operand lane layout (ISA 7.12.2):
//   lane half (lane>=16) adds +8 to K; per lane the 16 elements are two
//   contiguous K-runs of 8:  [half*8 .. half*8+7] and [16+half*8 .. +7].
// A (row-major) loads as two b128 per lane.  B-side is pre-packed (or
// LDS-staged) so each lane reads 32 contiguous bytes.
// Packed-B slot for source element (kk in 0..31, n):
//   lane = (n & 15) + 16*((kk>>3)&1) ;  e = (kk & 7) + 8*(kk>>4)
// ---------------------------------------------------------------------------

__device__ __forceinline__ v16bf load_a_b128(const __bf16* row, int k0, int half) {
    union { uint4 q[2]; v16bf v; } u;
    u.q[0] = *(const uint4*)(row + k0 + half * 8);
    u.q[1] = *(const uint4*)(row + k0 + 16 + half * 8);
    return u.v;
}

__device__ __forceinline__ v16bf load_packed_b(const __bf16* pk, size_t tile, int lane) {
    const uint4* p = (const uint4*)(pk + (tile * 32 + lane) * 16);
    union { uint4 q[2]; v16bf v; } u;
    u.q[0] = p[0];
    u.q[1] = p[1];
    return u.v;
}

__device__ __forceinline__ v16bf lds_read_b(const unsigned int* base) {
    union { uint4 q[2]; v16bf v; } u;
    u.q[0] = *(const uint4*)(base);
    u.q[1] = *(const uint4*)(base + 4);
    return u.v;
}

__device__ __forceinline__ unsigned int pack2_bf16(float lo, float hi) {
    union { __bf16 h; unsigned short u; } a, b;
    a.h = (__bf16)lo;
    b.h = (__bf16)hi;
    return (unsigned int)a.u | ((unsigned int)b.u << 16);
}

#define WMMA_BF16(A, Bm, C) \
    __builtin_amdgcn_wmma_f32_16x16x32_bf16(false, (A), false, (Bm), (short)0, (C), false, false)

// ---------------------------------------------------------------------------
// Reductions (wave32)
// ---------------------------------------------------------------------------
__device__ __forceinline__ float wave_sum_f(float v) {
#pragma unroll
    for (int off = 16; off > 0; off >>= 1) v += __shfl_xor(v, off, 32);
    return v;
}

__device__ __forceinline__ void block_sum2(float& s, float& s2, float* sm) {
    const int wid  = threadIdx.x >> 5;
    const int lane = threadIdx.x & 31;
    s  = wave_sum_f(s);
    s2 = wave_sum_f(s2);
    if (lane == 0) { sm[wid] = s; sm[wid + 8] = s2; }
    __syncthreads();
    if (wid == 0) {
        float a = (lane < 8) ? sm[lane]     : 0.0f;
        float b = (lane < 8) ? sm[lane + 8] : 0.0f;
        a = wave_sum_f(a);
        b = wave_sum_f(b);
        if (lane == 0) { sm[16] = a; sm[17] = b; }
    }
    __syncthreads();
    s = sm[16]; s2 = sm[17];
}

// ---------------------------------------------------------------------------
// 1) kin = x + LN(y); xr = per-128 LN(kin) with kn1 -> bf16 [B,N,128]
// ---------------------------------------------------------------------------
__global__ __launch_bounds__(256) void prep_kernel(
    const float* __restrict__ x, const float* __restrict__ y,
    const float* __restrict__ ny_g, const float* __restrict__ ny_b,
    const float* __restrict__ kn1_g, const float* __restrict__ kn1_b,
    __bf16* __restrict__ xr) {
    __shared__ float sm[18];
    const int row = blockIdx.x;                  // b*L + l
    const int t   = threadIdx.x;
    const float* yr  = y + (size_t)row * D_;
    const float* xrw = x + (size_t)row * D_;

    float yv[4];
    float s = 0.f, s2 = 0.f;
#pragma unroll
    for (int i = 0; i < 4; ++i) {
        yv[i] = yr[t * 4 + i];
        s += yv[i]; s2 += yv[i] * yv[i];
    }
    block_sum2(s, s2, sm);
    const float m1  = s * (1.0f / D_);
    const float rs1 = rsqrtf(s2 * (1.0f / D_) - m1 * m1 + 1e-5f);

    float kin[4];
    float bs = 0.f, bs2 = 0.f;
#pragma unroll
    for (int i = 0; i < 4; ++i) {
        const int d = t * 4 + i;
        const float ny = (yv[i] - m1) * rs1 * ny_g[d] + ny_b[d];
        kin[i] = xrw[d] + ny;
        bs += kin[i]; bs2 += kin[i] * kin[i];
    }
    bs  = wave_sum_f(bs);   // each wave == one 128-block
    bs2 = wave_sum_f(bs2);
    const float m2  = bs * (1.0f / BLK_);
    const float rs2 = rsqrtf(bs2 * (1.0f / BLK_) - m2 * m2 + 1e-5f);
#pragma unroll
    for (int i = 0; i < 4; ++i) {
        const int d = t * 4 + i;
        const int k = d & (BLK_ - 1);
        xr[(size_t)row * D_ + d] =
            (__bf16)((kin[i] - m2) * rs2 * kn1_g[k] + kn1_b[k]);
    }
}

// ---------------------------------------------------------------------------
// 2) Wn packed-B: per n (grid N_), 128x128 matrix, column-normalized, written
//    directly in WMMA-B lane layout (tiles of 32x16 -> 32 lanes x 16 bf16).
// ---------------------------------------------------------------------------
__global__ __launch_bounds__(128) void wprep_kernel(
    const float* __restrict__ W_1, const float* __restrict__ W_V,
    __bf16* __restrict__ Wnp) {
    const int n = blockIdx.x;
    const int o = threadIdx.x;                 // output column 0..127
    float w1[BASIS_];
#pragma unroll
    for (int k = 0; k < BASIS_; ++k) w1[k] = W_1[n * BASIS_ + k];

    float sq = 0.f;
    for (int i = 0; i < BLK_; ++i) {
        float acc = 0.f;
#pragma unroll
        for (int k = 0; k < BASIS_; ++k)
            acc = fmaf(w1[k], W_V[((size_t)k * BLK_ + i) * BLK_ + o], acc);
        sq = fmaf(acc, acc, sq);
    }
    const float scale = 1.0f / fmaxf(sqrtf(sq), 1e-12f);

    __bf16* base = Wnp + (size_t)n * (BLK_ * BLK_);
    const int nt = o >> 4;
    for (int i = 0; i < BLK_; ++i) {
        float acc = 0.f;
#pragma unroll
        for (int k = 0; k < BASIS_; ++k)
            acc = fmaf(w1[k], W_V[((size_t)k * BLK_ + i) * BLK_ + o], acc);
        const int kt = i >> 5, kk = i & 31;
        const int lm = (o & 15) + 16 * ((kk >> 3) & 1);
        const int e  = (kk & 7) + 8 * (kk >> 4);
        base[((size_t)(kt * 8 + nt) * 32 + lm) * 16 + e] = (__bf16)(acc * scale);
    }
}

// ---------------------------------------------------------------------------
// 3) Abf packed-B: column-normalized A (256x256) in WMMA-B lane layout.
// ---------------------------------------------------------------------------
__global__ __launch_bounds__(256) void aprep_kernel(
    const float* __restrict__ A_logits, __bf16* __restrict__ Abfp) {
    __shared__ float sm[18];
    const int j = blockIdx.x;
    const int i = threadIdx.x;
    const float v = A_logits[(size_t)i * N_ + j];
    float s = v * v, s2 = 0.f;
    block_sum2(s, s2, sm);
    const float scale = 1.0f / fmaxf(sqrtf(s), 1e-12f);
    const int kt = i >> 5, kk = i & 31;
    const int nt = j >> 4;
    const int lm = (j & 15) + 16 * ((kk >> 3) & 1);
    const int e  = (kk & 7) + 8 * (kk >> 4);
    Abfp[((size_t)(kt * 16 + nt) * 32 + lm) * 16 + e] = (__bf16)(v * scale);
}

// ---------------------------------------------------------------------------
// 4) x_local[b,n,:] = xr[b,n,:] @ Wn[n]  (M=64, K=128, Nt=128 per n)
// ---------------------------------------------------------------------------
__global__ __launch_bounds__(128) void xlocal_kernel(
    const __bf16* __restrict__ xr, const __bf16* __restrict__ Wnp,
    float* __restrict__ xl) {
    const int lane = threadIdx.x & 31;
    const int wid  = threadIdx.x >> 5;
    const int half = lane >> 4;
    const int mn   = lane & 15;
    const int nt0  = blockIdx.x * 64 + wid * 16;
    const int m0   = blockIdx.y * 16;
    const int n    = blockIdx.z;
    const __bf16* Bp   = Wnp + (size_t)n * (BLK_ * BLK_);
    const __bf16* Arow = xr + ((size_t)(m0 + mn) * N_ + n) * BLK_;

    v8f acc = {};
    for (int k0 = 0; k0 < BLK_; k0 += 32) {
        v16bf a = load_a_b128(Arow, k0, half);
        v16bf b = load_packed_b(Bp, (size_t)(k0 >> 5) * 8 + (nt0 >> 4), lane);
        acc = WMMA_BF16(a, b, acc);
    }
#pragma unroll
    for (int r = 0; r < 8; ++r) {
        const int m = m0 + r + half * 8;
        xl[((size_t)m * N_ + n) * BLK_ + nt0 + mn] = acc[r];
    }
}

// ---------------------------------------------------------------------------
// 5) xt[b,k,:] = LN_n(x_local[b,:,k]) with kn2 -> bf16 [B,128,256]
// ---------------------------------------------------------------------------
__global__ __launch_bounds__(256) void kn2_kernel(
    const float* __restrict__ xl, const float* __restrict__ kn2_g,
    const float* __restrict__ kn2_b, __bf16* __restrict__ xt) {
    __shared__ float sm[18];
    const int b  = blockIdx.x >> 7;
    const int kk = blockIdx.x & 127;
    const int n  = threadIdx.x;
    const float v = xl[(size_t)b * (N_ * BLK_) + (size_t)n * BLK_ + kk];
    float s = v, s2 = v * v;
    block_sum2(s, s2, sm);
    const float m  = s * (1.0f / N_);
    const float rs = rsqrtf(s2 * (1.0f / N_) - m * m + 1e-5f);
    xt[(size_t)b * (BLK_ * N_) + (size_t)kk * N_ + n] =
        (__bf16)((v - m) * rs * kn2_g[n] + kn2_b[n]);
}

// ---------------------------------------------------------------------------
// 6) x_global[b] = xt[b] @ Abf   (M=128, K=256, Nt=256) per batch b
// ---------------------------------------------------------------------------
__global__ __launch_bounds__(128) void xglobal_kernel(
    const __bf16* __restrict__ xt, const __bf16* __restrict__ Abfp,
    float* __restrict__ xg) {
    const int lane = threadIdx.x & 31;
    const int wid  = threadIdx.x >> 5;
    const int half = lane >> 4;
    const int mn   = lane & 15;
    const int n0   = blockIdx.x * 64 + wid * 16;
    const int m0   = blockIdx.y * 16;
    const int b    = blockIdx.z;
    const __bf16* Arow = xt + (size_t)b * (BLK_ * N_) + (size_t)(m0 + mn) * N_;

    v8f acc = {};
    for (int k0 = 0; k0 < N_; k0 += 32) {
        v16bf a  = load_a_b128(Arow, k0, half);
        v16bf bm = load_packed_b(Abfp, (size_t)(k0 >> 5) * 16 + (n0 >> 4), lane);
        acc = WMMA_BF16(a, bm, acc);
    }
#pragma unroll
    for (int r = 0; r < 8; ++r) {
        const int m = m0 + r + half * 8;
        xg[(size_t)b * (BLK_ * N_) + (size_t)m * N_ + n0 + mn] = acc[r];
    }
}

// ---------------------------------------------------------------------------
// 7) mixed = LN(x + 0.5*rule + 0.5*kron, n1); f32 [B,L,D] + bf16 [L,B,D]
// ---------------------------------------------------------------------------
__global__ __launch_bounds__(256) void mixed_kernel(
    const float* __restrict__ x, const float* __restrict__ xg,
    const float* __restrict__ n1_g, const float* __restrict__ n1_b,
    float* __restrict__ mixed, __bf16* __restrict__ mixedT) {
    __shared__ float sm[18];
    const int row = blockIdx.x;          // b*L + l
    const int b = row >> 5, l = row & 31;
    const int t = threadIdx.x;

    float v[4];
    float s = 0.f, s2 = 0.f;
#pragma unroll
    for (int i = 0; i < 4; ++i) {
        const int d = t * 4 + i;
        const int j = d >> 5, c = d & 31;
        const float xv = x[(size_t)row * D_ + d];
        const float rl = x[((size_t)b * L_ + j) * D_ + l * 32 + c];
        const float kr = xg[(size_t)b * (L_ * D_) + (size_t)l * D_ + d];
        v[i] = xv + 0.5f * rl + 0.5f * kr;
        s += v[i]; s2 += v[i] * v[i];
    }
    block_sum2(s, s2, sm);
    const float m  = s * (1.0f / D_);
    const float rs = rsqrtf(s2 * (1.0f / D_) - m * m + 1e-5f);
#pragma unroll
    for (int i = 0; i < 4; ++i) {
        const int d = t * 4 + i;
        const float o = (v[i] - m) * rs * n1_g[d] + n1_b[d];
        mixed[(size_t)row * D_ + d] = o;
        mixedT[((size_t)l * B_ + b) * D_ + d] = (__bf16)o;
    }
}

// ---------------------------------------------------------------------------
// 8) gate/up GEMMs + fused SiLU*up -> h (bf16 [L,B,DFF]).
//    Single-pass weight stream (768 MB roofline), software-pipelined:
//    global f32 loads of tile i+1 issue before WMMAs of tile i, LDS
//    ping-pong gives one barrier per K-step.  grid (DFF/64, 1, L), block 128.
// ---------------------------------------------------------------------------
__global__ __launch_bounds__(128) void gateup_kernel(
    const __bf16* __restrict__ mixedT, const float* __restrict__ Wg,
    const float* __restrict__ Wu, __bf16* __restrict__ h) {
    __shared__ __align__(16) unsigned int ldsG[2][1024];   // 2 x 4 KB ping-pong
    __shared__ __align__(16) unsigned int ldsU[2][1024];
    const int t    = threadIdx.x;
    const int lane = t & 31;
    const int wid  = t >> 5;
    const int half = lane >> 4;
    const int mn   = lane & 15;
    const int nblk = blockIdx.x * 64;
    const int l    = blockIdx.z;

    const __bf16* Abase = mixedT + (size_t)l * B_ * D_;
    const float*  Gb = Wg + (size_t)l * D_ * DFF_;
    const float*  Ub = Wu + (size_t)l * D_ * DFF_;

    // loop-invariant per-thread tile geometry (8 element-pairs per thread)
    size_t gofs[8];
    int    dofs[8];
#pragma unroll
    for (int i = 0; i < 8; ++i) {
        const int idx = t + i * 128;        // 0..1023
        const int c   = idx & 63;
        const int k   = (idx >> 6) << 1;    // even row 0..30
        gofs[i] = (size_t)k * DFF_ + nblk + c;
        const int w  = c >> 4;
        const int lm = (c & 15) + 16 * ((k >> 3) & 1);
        const int e2 = ((k & 7) >> 1) + ((k >> 4) << 2);
        dofs[i] = (w * 32 + lm) * 8 + e2;
    }
    const unsigned int rdo = (wid * 32 + lane) * 8;

    float gv[16], uv[16];
#pragma unroll
    for (int i = 0; i < 8; ++i) {           // prologue: tile 0 -> regs
        gv[2 * i] = Gb[gofs[i]];  gv[2 * i + 1] = Gb[gofs[i] + DFF_];
        uv[2 * i] = Ub[gofs[i]];  uv[2 * i + 1] = Ub[gofs[i] + DFF_];
    }
#pragma unroll
    for (int i = 0; i < 8; ++i) {           // pack tile 0 -> LDS buf 0
        ldsG[0][dofs[i]] = pack2_bf16(gv[2 * i], gv[2 * i + 1]);
        ldsU[0][dofs[i]] = pack2_bf16(uv[2 * i], uv[2 * i + 1]);
    }
    __syncthreads();

    v8f accg[4] = {}, accu[4] = {};
    for (int it = 0; it < D_ / 32; ++it) {
        const int cur = it & 1;
        if (it + 1 < D_ / 32) {             // issue next tile's loads early
            const size_t kofs = (size_t)(it + 1) * 32 * DFF_;
#pragma unroll
            for (int i = 0; i < 8; ++i) {
                gv[2 * i] = Gb[kofs + gofs[i]];  gv[2 * i + 1] = Gb[kofs + gofs[i] + DFF_];
                uv[2 * i] = Ub[kofs + gofs[i]];  uv[2 * i + 1] = Ub[kofs + gofs[i] + DFF_];
            }
        }
        const v16bf bg = lds_read_b(&ldsG[cur][rdo]);
        const v16bf bu = lds_read_b(&ldsU[cur][rdo]);
#pragma unroll
        for (int mt = 0; mt < 4; ++mt) {
            const v16bf a = load_a_b128(Abase + (size_t)(mt * 16 + mn) * D_, it * 32, half);
            accg[mt] = WMMA_BF16(a, bg, accg[mt]);
            accu[mt] = WMMA_BF16(a, bu, accu[mt]);
        }
        if (it + 1 < D_ / 32) {             // pack next tile into other buffer
#pragma unroll
            for (int i = 0; i < 8; ++i) {
                ldsG[cur ^ 1][dofs[i]] = pack2_bf16(gv[2 * i], gv[2 * i + 1]);
                ldsU[cur ^ 1][dofs[i]] = pack2_bf16(uv[2 * i], uv[2 * i + 1]);
            }
        }
        __syncthreads();
    }

    const int n = nblk + wid * 16 + mn;
#pragma unroll
    for (int mt = 0; mt < 4; ++mt) {
#pragma unroll
        for (int r = 0; r < 8; ++r) {
            const int m = mt * 16 + r + half * 8;
            const float g  = accg[mt][r];
            const float sw = g / (1.0f + __expf(-g));
            h[((size_t)l * B_ + m) * DFF_ + n] = (__bf16)(sw * accu[mt][r]);
        }
    }
}

// ---------------------------------------------------------------------------
// 9) ffn = h @ W_down  (M=64, K=2048, Nt=1024) per token -> f32 [B,L,D]
//    Same pipelined single-pass weight stream.  grid (D/64, 1, L), block 128.
// ---------------------------------------------------------------------------
__global__ __launch_bounds__(128) void down_kernel(
    const __bf16* __restrict__ h, const float* __restrict__ Wd,
    float* __restrict__ ffn) {
    __shared__ __align__(16) unsigned int ldsB[2][1024];
    const int t    = threadIdx.x;
    const int lane = t & 31;
    const int wid  = t >> 5;
    const int half = lane >> 4;
    const int mn   = lane & 15;
    const int nblk = blockIdx.x * 64;
    const int l    = blockIdx.z;

    const __bf16* Abase = h + (size_t)l * B_ * DFF_;
    const float*  Db = Wd + (size_t)l * DFF_ * D_;

    size_t gofs[8];
    int    dofs[8];
#pragma unroll
    for (int i = 0; i < 8; ++i) {
        const int idx = t + i * 128;
        const int c   = idx & 63;
        const int k   = (idx >> 6) << 1;
        gofs[i] = (size_t)k * D_ + nblk + c;
        const int w  = c >> 4;
        const int lm = (c & 15) + 16 * ((k >> 3) & 1);
        const int e2 = ((k & 7) >> 1) + ((k >> 4) << 2);
        dofs[i] = (w * 32 + lm) * 8 + e2;
    }
    const unsigned int rdo = (wid * 32 + lane) * 8;

    float bv[16];
#pragma unroll
    for (int i = 0; i < 8; ++i) {
        bv[2 * i] = Db[gofs[i]];  bv[2 * i + 1] = Db[gofs[i] + D_];
    }
#pragma unroll
    for (int i = 0; i < 8; ++i)
        ldsB[0][dofs[i]] = pack2_bf16(bv[2 * i], bv[2 * i + 1]);
    __syncthreads();

    v8f acc[4] = {};
    for (int it = 0; it < DFF_ / 32; ++it) {
        const int cur = it & 1;
        if (it + 1 < DFF_ / 32) {
            const size_t kofs = (size_t)(it + 1) * 32 * D_;
#pragma unroll
            for (int i = 0; i < 8; ++i) {
                bv[2 * i] = Db[kofs + gofs[i]];  bv[2 * i + 1] = Db[kofs + gofs[i] + D_];
            }
        }
        const v16bf bm = lds_read_b(&ldsB[cur][rdo]);
#pragma unroll
        for (int mt = 0; mt < 4; ++mt) {
            const v16bf a = load_a_b128(Abase + (size_t)(mt * 16 + mn) * DFF_, it * 32, half);
            acc[mt] = WMMA_BF16(a, bm, acc[mt]);
        }
        if (it + 1 < DFF_ / 32) {
#pragma unroll
            for (int i = 0; i < 8; ++i)
                ldsB[cur ^ 1][dofs[i]] = pack2_bf16(bv[2 * i], bv[2 * i + 1]);
        }
        __syncthreads();
    }

    const int n = nblk + wid * 16 + mn;
#pragma unroll
    for (int mt = 0; mt < 4; ++mt) {
#pragma unroll
        for (int r = 0; r < 8; ++r) {
            const int m = mt * 16 + r + half * 8;
            ffn[((size_t)m * L_ + l) * D_ + n] = acc[mt][r];
        }
    }
}

// ---------------------------------------------------------------------------
// 10) out = LN(mixed + ffn, n2); y_out = y + ffn
// ---------------------------------------------------------------------------
__global__ __launch_bounds__(256) void final_kernel(
    const float* __restrict__ mixed, const float* __restrict__ ffn,
    const float* __restrict__ y, const float* __restrict__ n2_g,
    const float* __restrict__ n2_b, float* __restrict__ out) {
    __shared__ float sm[18];
    const int row = blockIdx.x;
    const int t = threadIdx.x;
    float v[4], f[4];
    float s = 0.f, s2 = 0.f;
#pragma unroll
    for (int i = 0; i < 4; ++i) {
        const int d = t * 4 + i;
        f[i] = ffn[(size_t)row * D_ + d];
        v[i] = mixed[(size_t)row * D_ + d] + f[i];
        s += v[i]; s2 += v[i] * v[i];
    }
    block_sum2(s, s2, sm);
    const float m  = s * (1.0f / D_);
    const float rs = rsqrtf(s2 * (1.0f / D_) - m * m + 1e-5f);
#pragma unroll
    for (int i = 0; i < 4; ++i) {
        const int d = t * 4 + i;
        out[(size_t)row * D_ + d] = (v[i] - m) * rs * n2_g[d] + n2_b[d];
        out[BLD_ + (size_t)row * D_ + d] = y[(size_t)row * D_ + d] + f[i];
    }
}

// ---------------------------------------------------------------------------
// Launcher
// ---------------------------------------------------------------------------
extern "C" void kernel_launch(void* const* d_in, const int* in_sizes, int n_in,
                              void* d_out, int out_size, void* d_ws, size_t ws_size,
                              hipStream_t stream) {
    (void)in_sizes; (void)n_in; (void)out_size; (void)ws_size;
    const float* x        = (const float*)d_in[0];
    const float* y        = (const float*)d_in[1];
    const float* A_logits = (const float*)d_in[2];
    const float* W_1      = (const float*)d_in[3];
    const float* W_V      = (const float*)d_in[4];
    const float* kn1_g    = (const float*)d_in[5];
    const float* kn1_b    = (const float*)d_in[6];
    const float* kn2_g    = (const float*)d_in[7];
    const float* kn2_b    = (const float*)d_in[8];
    const float* W_gate   = (const float*)d_in[9];
    const float* W_up     = (const float*)d_in[10];
    const float* W_down   = (const float*)d_in[11];
    const float* ny_g     = (const float*)d_in[12];
    const float* ny_b     = (const float*)d_in[13];
    const float* n1_g     = (const float*)d_in[14];
    const float* n1_b     = (const float*)d_in[15];
    const float* n2_g     = (const float*)d_in[16];
    const float* n2_b     = (const float*)d_in[17];
    float* out = (float*)d_out;

    char* ws = (char*)d_ws;
    size_t off = 0;
    __bf16* xr     = (__bf16*)(ws + off); off += (size_t)B_ * N_ * BLK_ * 2;   // 4 MiB
    __bf16* Wnp    = (__bf16*)(ws + off); off += (size_t)N_ * BLK_ * BLK_ * 2; // 8 MiB (packed B)
    __bf16* Abfp   = (__bf16*)(ws + off); off += (size_t)N_ * N_ * 2;          // 128 KiB (packed B)
    float*  xl     = (float*) (ws + off); off += (size_t)B_ * N_ * BLK_ * 4;   // 8 MiB
    __bf16* xt     = (__bf16*)(ws + off); off += (size_t)B_ * BLK_ * N_ * 2;   // 4 MiB
    float*  xg     = (float*) (ws + off); off += (size_t)B_ * BLK_ * N_ * 4;   // 8 MiB
    float*  mixed  = (float*) (ws + off); off += BLD_ * 4;                     // 8 MiB
    __bf16* mixedT = (__bf16*)(ws + off); off += BLD_ * 2;                     // 4 MiB
    __bf16* hbuf   = (__bf16*)(ws + off); off += (size_t)L_ * B_ * DFF_ * 2;   // 8 MiB
    float*  ffn    = (float*) (ws + off); off += BLD_ * 4;                     // 8 MiB

    prep_kernel   <<<dim3(B_ * L_),          256, 0, stream>>>(x, y, ny_g, ny_b, kn1_g, kn1_b, xr);
    wprep_kernel  <<<dim3(N_),               128, 0, stream>>>(W_1, W_V, Wnp);
    aprep_kernel  <<<dim3(N_),               256, 0, stream>>>(A_logits, Abfp);
    xlocal_kernel <<<dim3(2, 4, N_),         128, 0, stream>>>(xr, Wnp, xl);
    kn2_kernel    <<<dim3(B_ * BLK_),        256, 0, stream>>>(xl, kn2_g, kn2_b, xt);
    xglobal_kernel<<<dim3(4, 8, B_),         128, 0, stream>>>(xt, Abfp, xg);
    mixed_kernel  <<<dim3(B_ * L_),          256, 0, stream>>>(x, xg, n1_g, n1_b, mixed, mixedT);
    gateup_kernel <<<dim3(DFF_ / 64, 1, L_), 128, 0, stream>>>(mixedT, W_gate, W_up, hbuf);
    down_kernel   <<<dim3(D_ / 64, 1, L_),   128, 0, stream>>>(hbuf, W_down, ffn);
    final_kernel  <<<dim3(B_ * L_),          256, 0, stream>>>(mixed, ffn, y, n2_g, n2_b, out);
}
